// GraphAttentionLayer_62577673503434
// MI455X (gfx1250) — compile-verified, matrix operand order checked by reading, not run on hardware
//
#include <hip/hip_runtime.h>

typedef __attribute__((ext_vector_type(16))) _Float16 v16h;
typedef __attribute__((ext_vector_type(8)))  _Float16 v8h;
typedef __attribute__((ext_vector_type(8)))  float    v8f;

static constexpr int Bn = 8;
static constexpr int Nn = 2048;
static constexpr int Fo = 128;
static constexpr float ALPHA = 0.2f;

#define WMMA_F16(A, B, C) \
  __builtin_amdgcn_wmma_f32_16x16x32_f16(false, (A), false, (B), (short)0, (C), false, false)

// ---------------------------------------------------------------------------
// 1) Pack adj (int32 0/1) into bitmask: bits[i][w] bit l == adj[i][w*32+l]>0
// ---------------------------------------------------------------------------
__global__ void pack_adj_kernel(const int* __restrict__ adj,
                                unsigned* __restrict__ bits) {
  int idx = blockIdx.x * blockDim.x + threadIdx.x;   // 0 .. N*64-1
  int i = idx >> 6;
  int w = idx & 63;
  const int* row = adj + (size_t)i * Nn + w * 32;
  unsigned m = 0u;
#pragma unroll
  for (int t = 0; t < 32; ++t) m |= (row[t] > 0 ? 1u : 0u) << t;
  bits[idx] = m;
}

// ---------------------------------------------------------------------------
// 2) Wh = h @ W via WMMA f16; store transposed f16: WhT[b][f][j]
//    Block: 256 thr (8 waves), covers 128 rows x 128 fout for one batch.
// ---------------------------------------------------------------------------
__global__ void __launch_bounds__(256)
wh_kernel(const float* __restrict__ h, const float* __restrict__ W,
          _Float16* __restrict__ WhT) {
  __shared__ _Float16 smem[128 * 136];   // phase1: WT[fout][fin]; phase2: tileT[f][jrel]

  const int b  = blockIdx.x >> 4;
  const int i0 = (blockIdx.x & 15) * 128;
  const int tid  = threadIdx.x;
  const int lane = tid & 31;
  const int wave = tid >> 5;
  const int hs   = lane >> 4;       // half-select (K-group)
  const int mr   = lane & 15;

  // stage W transposed into LDS as f16, pitch 136 halves (bank-safe)
  for (int idx = tid; idx < 128 * 128; idx += 256) {
    int fin = idx >> 7, fo = idx & 127;
    smem[fo * 136 + fin] = (_Float16)W[fin * 128 + fo];
  }
  __syncthreads();

  const int row = i0 + wave * 16 + mr;
  const float* hrow = h + ((size_t)b * Nn + row) * Fo;

  v8f acc[8] = {};
#pragma unroll
  for (int c = 0; c < 4; ++c) {
    const int cb = c * 32;
    // ---- A fragment: h rows, f32 -> f16, per ISA 16-bit A layout
    v16h a;
    {
      const float* p = hrow + cb + hs * 8;
      float4 p0 = *(const float4*)(p);
      float4 p1 = *(const float4*)(p + 4);
      float4 q0 = *(const float4*)(p + 16);
      float4 q1 = *(const float4*)(p + 20);
      a[0]=(_Float16)p0.x; a[1]=(_Float16)p0.y; a[2]=(_Float16)p0.z; a[3]=(_Float16)p0.w;
      a[4]=(_Float16)p1.x; a[5]=(_Float16)p1.y; a[6]=(_Float16)p1.z; a[7]=(_Float16)p1.w;
      a[8]=(_Float16)q0.x; a[9]=(_Float16)q0.y; a[10]=(_Float16)q0.z; a[11]=(_Float16)q0.w;
      a[12]=(_Float16)q1.x; a[13]=(_Float16)q1.y; a[14]=(_Float16)q1.z; a[15]=(_Float16)q1.w;
    }
#pragma unroll
    for (int n = 0; n < 8; ++n) {
      const _Float16* wr = smem + (n * 16 + mr) * 136 + cb + hs * 8;
      v8h x = *(const v8h*)wr;
      v8h y = *(const v8h*)(wr + 16);
      v16h bf = __builtin_shufflevector(x, y, 0,1,2,3,4,5,6,7,8,9,10,11,12,13,14,15);
      acc[n] = WMMA_F16(a, bf, acc[n]);
    }
  }

  __syncthreads();  // done reading WT; reuse smem as transpose tile
  // D layout: VGPR v -> row M = v + hs*8 (lane group), col N = lane&15
#pragma unroll
  for (int n = 0; n < 8; ++n) {
    const int f = n * 16 + mr;
#pragma unroll
    for (int v = 0; v < 8; ++v) {
      int jrel = wave * 16 + v + hs * 8;
      smem[f * 136 + jrel] = (_Float16)acc[n][v];
    }
  }
  __syncthreads();

  // coalesced write of WhT[b][f][i0 .. i0+127]
  for (int k = 0; k < 8; ++k) {
    int seg  = tid + k * 256;          // 2048 16-byte segments
    int f    = seg >> 4;
    int jseg = (seg & 15) * 8;
    v8h val = *(const v8h*)(smem + f * 136 + jseg);
    *(v8h*)(WhT + ((size_t)b * Fo + f) * Nn + i0 + jseg) = val;
  }
}

// ---------------------------------------------------------------------------
// 3) s1[b][j] = Wh[b][j][:].a1 ; s2 likewise (column reads from WhT, coalesced)
// ---------------------------------------------------------------------------
__global__ void s_kernel(const _Float16* __restrict__ WhT,
                         const float* __restrict__ a,
                         float* __restrict__ s1, float* __restrict__ s2) {
  int idx = blockIdx.x * 256 + threadIdx.x;   // 0..B*N-1
  int b = idx >> 11, j = idx & (Nn - 1);
  const _Float16* base = WhT + (size_t)b * Fo * Nn + j;
  float r1 = 0.f, r2 = 0.f;
#pragma unroll 8
  for (int f = 0; f < Fo; ++f) {
    float wv = (float)base[(size_t)f * Nn];
    r1 += wv * a[f];
    r2 += wv * a[Fo + f];
  }
  s1[idx] = r1;
  s2[idx] = r2;
}

// ---------------------------------------------------------------------------
// 4) Per-row softmax stats: m[b][i] = max_j masked lrelu(s1+s2); invden = 1/sum
//    One wave32 per row; shuffle reductions.
// ---------------------------------------------------------------------------
__global__ void stat_kernel(const float* __restrict__ s1,
                            const float* __restrict__ s2,
                            const unsigned* __restrict__ bits,
                            float* __restrict__ mrow,
                            float* __restrict__ invden) {
  int tid  = threadIdx.x;
  int lane = tid & 31;
  int w = blockIdx.x * 8 + (tid >> 5);     // row id, 0..B*N-1
  int b = w >> 11, i = w & (Nn - 1);
  const float* s2b = s2 + (b << 11);
  const unsigned* br = bits + (size_t)i * 64;
  float s1i = s1[w];

  float mx = -3.0e38f;
  for (int c = 0; c < 64; ++c) {
    unsigned word = br[c];
    if ((word >> lane) & 1u) {
      float e = s1i + s2b[c * 32 + lane];
      e = e > 0.f ? e : ALPHA * e;
      mx = fmaxf(mx, e);
    }
  }
#pragma unroll
  for (int off = 16; off > 0; off >>= 1) mx = fmaxf(mx, __shfl_xor(mx, off, 32));

  float sum = 0.f;
  for (int c = 0; c < 64; ++c) {
    unsigned word = br[c];
    if ((word >> lane) & 1u) {
      float e = s1i + s2b[c * 32 + lane];
      e = e > 0.f ? e : ALPHA * e;
      sum += __expf(e - mx);
    }
  }
#pragma unroll
  for (int off = 16; off > 0; off >>= 1) sum += __shfl_xor(sum, off, 32);

  if (lane == 0) {
    mrow[w]   = mx;
    invden[w] = sum > 0.f ? 1.f / sum : 0.f;
  }
}

// ---------------------------------------------------------------------------
// 5) h_prime = softmax(e) @ Wh, flash-style.
//    Per wave: 16 rows x 128 fout. K (=j) chunks of 32.
//    B-tiles (WhT[b][:, j0..j0+31], 8KB) are staged ONCE per block into a
//    double-buffered LDS tile via CDNA5 async load-to-LDS (ASYNCcnt),
//    overlapping the DMA of chunk c+1 with the exp/WMMA compute of chunk c.
// ---------------------------------------------------------------------------
static constexpr int TPITCH = 56;   // halves; 112B row pitch: 16B-aligned,
                                    // 28-dword lane stride -> bank-conflict-free

__global__ void __launch_bounds__(256)
attn_kernel(const _Float16* __restrict__ WhT,
            const float* __restrict__ s1, const float* __restrict__ s2f,
            const unsigned* __restrict__ bits,
            const float* __restrict__ mrow, const float* __restrict__ invden,
            float* __restrict__ out) {
  __shared__ __align__(16) _Float16 tile[2][128 * TPITCH];

  const int b  = blockIdx.x >> 4;
  const int i0 = (blockIdx.x & 15) * 128;
  const int tid  = threadIdx.x;
  const int lane = tid & 31;
  const int wave = tid >> 5;
  const int hs   = lane >> 4;
  const int mr   = lane & 15;

  const int row = i0 + wave * 16 + mr;       // this lane's A-matrix row
  const int w   = (b << 11) + row;
  const float s1r = s1[w];
  const float mR  = mrow[w];
  const float idR = invden[w];
  const unsigned* adjr = bits + (size_t)row * 64;
  const float* s2b = s2f + (b << 11);
  const _Float16* whb = WhT + (size_t)b * Fo * Nn;

  // issue async DMA of chunk cs's B-tile into tile[cs&1]
  // 512 segments of 16B (128 f-rows x 4), 2 per thread
  auto issue_stage = [&](int cs) {
    const int j0s = cs * 32;
    unsigned base = (unsigned)(uintptr_t)(&tile[cs & 1][0]);
#pragma unroll
    for (int k = 0; k < 2; ++k) {
      int seg  = tid + k * 256;
      int f    = seg >> 2;
      int part = seg & 3;
      unsigned long long gaddr =
          (unsigned long long)(uintptr_t)(whb + (size_t)f * Nn + j0s + part * 8);
      unsigned ldsoff = base + (unsigned)((f * TPITCH + part * 8) * 2);
      asm volatile("global_load_async_to_lds_b128 %0, %1, off"
                   :: "v"(ldsoff), "v"(gaddr)
                   : "memory");
    }
  };

  v8f acc[8] = {};

  issue_stage(0);
  for (int c = 0; c < 64; ++c) {
    const int j0 = c * 32;

    if (c < 63) {
      issue_stage(c + 1);                                // prefetch next tile
      asm volatile("s_wait_asynccnt 0x2" ::: "memory");  // chunk c's DMA done
    } else {
      asm volatile("s_wait_asynccnt 0x0" ::: "memory");
    }
    __syncthreads();                                     // tile[c&1] visible to all waves

    const _Float16* tcur = &tile[c & 1][0];
    const unsigned aw = adjr[c];

    // s2 values for this lane's K slots: {hs*8..hs*8+7} and {16+hs*8..+7}
    const float* sp = s2b + j0 + hs * 8;
    float4 p0 = *(const float4*)(sp);
    float4 p1 = *(const float4*)(sp + 4);
    float4 q0 = *(const float4*)(sp + 16);
    float4 q1 = *(const float4*)(sp + 20);
    float sv[16] = {p0.x, p0.y, p0.z, p0.w, p1.x, p1.y, p1.z, p1.w,
                    q0.x, q0.y, q0.z, q0.w, q1.x, q1.y, q1.z, q1.w};

    // A fragment: probabilities, built directly in ISA 16-bit A layout
    v16h a;
#pragma unroll
    for (int t = 0; t < 16; ++t) {
      int koff = (t < 8) ? (hs * 8 + t) : (8 + hs * 8 + t);  // j-offset in chunk
      float e = s1r + sv[t];
      e = e > 0.f ? e : ALPHA * e;
      float p = ((aw >> koff) & 1u) ? __expf(e - mR) * idR : 0.f;
      a[t] = (_Float16)p;
    }

#pragma unroll
    for (int n = 0; n < 8; ++n) {
      const _Float16* tp = tcur + (n * 16 + mr) * TPITCH + hs * 8;
      v8h x = *(const v8h*)tp;
      v8h y = *(const v8h*)(tp + 16);
      v16h bf = __builtin_shufflevector(x, y, 0,1,2,3,4,5,6,7,8,9,10,11,12,13,14,15);
      acc[n] = WMMA_F16(a, bf, acc[n]);
    }

    __syncthreads();   // all waves done with tile[c&1] before it is re-staged
  }

  // D: VGPR v -> row M = v + hs*8, col N = lane&15
#pragma unroll
  for (int n = 0; n < 8; ++n) {
#pragma unroll
    for (int v = 0; v < 8; ++v) {
      int r = i0 + wave * 16 + v + hs * 8;
      out[((size_t)b * Nn + r) * Fo + n * 16 + mr] = acc[n][v];
    }
  }
}

// ---------------------------------------------------------------------------
extern "C" void kernel_launch(void* const* d_in, const int* in_sizes, int n_in,
                              void* d_out, int out_size, void* d_ws, size_t ws_size,
                              hipStream_t stream) {
  const float* h   = (const float*)d_in[0];
  const int*   adj = (const int*)d_in[1];
  const float* W   = (const float*)d_in[2];
  const float* a   = (const float*)d_in[3];
  float* out = (float*)d_out;

  char* ws = (char*)d_ws;
  size_t off = 0;
  _Float16* WhT = (_Float16*)(ws + off); off += (size_t)Bn * Fo * Nn * sizeof(_Float16); // 4 MB
  float* s1   = (float*)(ws + off); off += (size_t)Bn * Nn * sizeof(float);
  float* s2   = (float*)(ws + off); off += (size_t)Bn * Nn * sizeof(float);
  float* mr   = (float*)(ws + off); off += (size_t)Bn * Nn * sizeof(float);
  float* invd = (float*)(ws + off); off += (size_t)Bn * Nn * sizeof(float);
  unsigned* bits = (unsigned*)(ws + off); off += (size_t)Nn * 64 * sizeof(unsigned);

  pack_adj_kernel<<<(Nn * 64) / 256, 256, 0, stream>>>(adj, bits);
  wh_kernel<<<Bn * (Nn / 128), 256, 0, stream>>>(h, W, WhT);
  s_kernel<<<(Bn * Nn) / 256, 256, 0, stream>>>(WhT, a, s1, s2);
  stat_kernel<<<(Bn * Nn) / 8, 256, 0, stream>>>(s1, s2, bits, mr, invd);
  attn_kernel<<<Bn * (Nn / 128), 256, 0, stream>>>(WhT, s1, s2, bits, mr, invd, out);
}
